// VectorQuantizer_12687333393030
// MI455X (gfx1250) — matrix operand order repaired and checked by reading, not compiled, not run on hardware
//
#include <hip/hip_runtime.h>
#include <hip/hip_bf16.h>

typedef __attribute__((ext_vector_type(16))) _Float16 v16h;
typedef __attribute__((ext_vector_type(8)))  float    v8f;

#define NUM_E    1024
#define DIM      64
#define NVEC     32768      // 8*64*64 vectors
#define SPATIAL  4096       // 64*64
#define NTILES   2048       // NVEC/16
#define OUT_Q_OFF    ((size_t)1)
#define OUT_PERP_OFF ((size_t)2097153)
#define OUT_ENC_OFF  ((size_t)2097154)

// ---- workspace layout (d_ws) ----
// uint   counts[1024]      @ byte 256
// float  enorm[1024]       @ float 2048 (byte 8192)
// f16    bpack[65536]      @ byte 16384 (128 KB, WMMA B fragments)
// float  partial[2048]     @ float 36864 (byte 147456)

// ---------------------------------------------------------------------------
// Pack codebook into WMMA B fragments in contiguous k order (16x16x32 f16, 32x16 KxN tiles):
// lane l (n = l&15, hi = l>>4), element e in v16h -> K = 16*hi + e.
// Also compute ||e_j||^2 in f32 and zero the code-usage histogram.
// ---------------------------------------------------------------------------
__global__ __launch_bounds__(256) void vq_prep(const float* __restrict__ emb,
                                               _Float16* __restrict__ bpack,
                                               float* __restrict__ enorm,
                                               unsigned* __restrict__ counts) {
  int j = blockIdx.x * 256 + threadIdx.x;   // code index 0..1023
  if (j >= NUM_E) return;
  int nt  = j >> 4;        // code tile
  int nin = j & 15;        // position within tile (the "N" lane)
  float nrm = 0.f;
  for (int d = 0; d < DIM; ++d) {
    float v = emb[j * DIM + d];
    nrm += v * v;
    int c    = d >> 5;          // which K=32 chunk
    int kk   = d & 31;          // K within chunk
    int hi2  = kk >> 4;         // lane half
    int lane = nin + (hi2 << 4);
    bpack[(((nt * 2 + c) * 32 + lane) << 4) + (kk & 15)] = (_Float16)v;
  }
  enorm[j]  = nrm;
  counts[j] = 0u;
}

// ---------------------------------------------------------------------------
// Zero the 128 MB one-hot encodings region (float2 stores, 8B aligned).
// ---------------------------------------------------------------------------
__global__ __launch_bounds__(256) void vq_zero(float2* __restrict__ enc, int n2) {
  int stride = gridDim.x * blockDim.x;
  float2 z; z.x = 0.f; z.y = 0.f;
  for (int i = blockIdx.x * blockDim.x + threadIdx.x; i < n2; i += stride)
    enc[i] = z;
}

// ---------------------------------------------------------------------------
// Main: one wave per 16-row A-tile. 64 code tiles x 2 WMMA(16x16x32 f16).
// dist = ||e||^2 - 2 x.e  (||x||^2 constant per row, irrelevant for argmin).
// ---------------------------------------------------------------------------
__global__ __launch_bounds__(256) void vq_main(const float* __restrict__ in,
                                               const float* __restrict__ emb,
                                               const _Float16* __restrict__ bpack,
                                               const float* __restrict__ enorm,
                                               float* __restrict__ out,
                                               float* __restrict__ partial,
                                               unsigned* __restrict__ counts) {
  __shared__ int s_idx[8][16];
  const int lane = threadIdx.x & 31;
  const int wave = threadIdx.x >> 5;
  const int tile = blockIdx.x * 8 + wave;    // 0..2047
  const int m    = lane & 15;
  const int hi   = lane >> 4;
  const int rowbase = tile << 4;             // first flat row of tile
  const int b       = rowbase >> 12;         // batch (4096 spatial per batch)
  const int sbase   = rowbase & 4095;        // spatial base
  const float* inb = in + (size_t)b * DIM * SPATIAL + sbase; // + d*4096 + m

  // A fragments: contiguous k order, same as bpack (not the ISA table order). lane (m,hi), element e ->
  // K = 16*hi + e; column d = 32*c + K; x at inb[d*4096 + m].
  v16h a[2];
#pragma unroll
  for (int c = 0; c < 2; ++c) {
#pragma unroll
    for (int e = 0; e < 16; ++e) {
      int k = (hi << 4) + e;
      int d = (c << 5) + k;
      a[c][e] = (_Float16)inb[(size_t)d * SPATIAL + m];
    }
  }

  const v16h* bpv = (const v16h*)bpack;
  float bestd[8];
  int   besti[8];
#pragma unroll
  for (int r = 0; r < 8; ++r) { bestd[r] = 3.0e38f; besti[r] = 0; }

  for (int nt = 0; nt < 64; ++nt) {
    v16h b0 = bpv[(nt * 2 + 0) * 32 + lane];
    v16h b1 = bpv[(nt * 2 + 1) * 32 + lane];
    if (nt < 63) __builtin_prefetch(&bpv[(nt + 1) * 2 * 32 + lane], 0, 0);
    v8f acc = {};
    acc = __builtin_amdgcn_wmma_f32_16x16x32_f16(false, a[0], false, b0,
                                                 (short)0, acc, false, false);
    acc = __builtin_amdgcn_wmma_f32_16x16x32_f16(false, a[1], false, b1,
                                                 (short)0, acc, false, false);
    // D layout: VGPR r, lanes 0-15 -> (M=r, N=m); lanes 16-31 -> (M=8+r, N=m)
    float en  = enorm[(nt << 4) + m];
    int   idx = (nt << 4) + m;
#pragma unroll
    for (int r = 0; r < 8; ++r) {
      float dist = en - 2.0f * acc[r];
      if (dist < bestd[r]) { bestd[r] = dist; besti[r] = idx; }
    }
  }

  // argmin across the 16 lanes of each half-wave (xor stays within halves)
#pragma unroll
  for (int mask = 8; mask >= 1; mask >>= 1) {
#pragma unroll
    for (int r = 0; r < 8; ++r) {
      float od = __shfl_xor(bestd[r], mask, 32);
      int   oi = __shfl_xor(besti[r], mask, 32);
      if (od < bestd[r] || (od == bestd[r] && oi < besti[r])) {
        bestd[r] = od; besti[r] = oi;
      }
    }
  }
  if (m == 0) {
#pragma unroll
    for (int r = 0; r < 8; ++r) s_idx[wave][(hi << 3) + r] = besti[r];
  }
  __syncthreads();

  // gather quantized, write NCHW output, accumulate MSE
  const int myidx = s_idx[wave][m];
  const float* qrow = emb + (size_t)myidx * DIM;
  float sse_l = 0.f;
#pragma unroll 8
  for (int dd = 0; dd < DIM; dd += 2) {
    int d   = dd + hi;
    float q = qrow[d];
    float x = inb[(size_t)d * SPATIAL + m];
    float t = q - x;
    sse_l += t * t;
    out[OUT_Q_OFF + (size_t)(b * DIM + d) * SPATIAL + sbase + m] = q;
  }

  // one-hot scatter + deterministic integer histogram
  if (hi == 0) {
    out[OUT_ENC_OFF + (size_t)(rowbase + m) * NUM_E + myidx] = 1.0f;
    atomicAdd(&counts[myidx], 1u);
  }

  // per-wave MSE partial (deterministic: no float atomics)
#pragma unroll
  for (int mask = 16; mask >= 1; mask >>= 1)
    sse_l += __shfl_xor(sse_l, mask, 32);
  if (lane == 0) partial[tile] = sse_l;
}

// ---------------------------------------------------------------------------
// Deterministic finalize: loss = 1.25 * MSE ; perplexity = exp(entropy).
// ---------------------------------------------------------------------------
__global__ __launch_bounds__(1024) void vq_finalize(const float* __restrict__ partial,
                                                    const unsigned* __restrict__ counts,
                                                    float* __restrict__ out) {
  __shared__ float rp[1024];
  __shared__ float re[1024];
  int j = threadIdx.x;
  rp[j] = partial[j] + partial[j + 1024];
  float p = (float)counts[j] * (1.0f / (float)NVEC);
  re[j] = p * logf(p + 1e-10f);
  __syncthreads();
  for (int s = 512; s > 0; s >>= 1) {
    if (j < s) { rp[j] += rp[j + s]; re[j] += re[j + s]; }
    __syncthreads();
  }
  if (j == 0) {
    out[0]            = 1.25f * rp[0] * (1.0f / 2097152.0f);
    out[OUT_PERP_OFF] = expf(-re[0]);
  }
}

extern "C" void kernel_launch(void* const* d_in, const int* in_sizes, int n_in,
                              void* d_out, int out_size, void* d_ws, size_t ws_size,
                              hipStream_t stream) {
  const float* in  = (const float*)d_in[0];   // [8,64,64,64] NCHW f32
  const float* emb = (const float*)d_in[1];   // [1024,64] f32
  float* out = (float*)d_out;

  unsigned* counts = (unsigned*)((char*)d_ws + 256);
  float*    enorm  = (float*)d_ws + 2048;
  _Float16* bpack  = (_Float16*)((char*)d_ws + 16384);
  float*    part   = (float*)d_ws + 36864;

  vq_prep<<<4, 256, 0, stream>>>(emb, bpack, enorm, counts);
  vq_zero<<<2048, 256, 0, stream>>>((float2*)(out + OUT_ENC_OFF),
                                    (NVEC * NUM_E) / 2);
  vq_main<<<NTILES / 8, 256, 0, stream>>>(in, emb, bpack, enorm, out, part, counts);
  vq_finalize<<<1, 1024, 0, stream>>>(part, counts, out);
}
